// cmMambaWithCNN_67980742361143
// MI455X (gfx1250) — matrix operand order repaired and checked by reading, not compiled, not run on hardware
//
#include <hip/hip_runtime.h>
#include <hip/hip_bf16.h>
#include <stdint.h>

// ---------------------------------------------------------------------------
// MambaWithCNN for MI455X (gfx1250, wave32, WMMA).
// Sizes: B=4, C=64, H=W=128, L=16384, BL=65536, D_INNER=128, D_STATE=16,
// DT_RANK=4, XPN=36.
// ---------------------------------------------------------------------------

typedef _Float16 h16_t;
typedef __attribute__((ext_vector_type(16))) _Float16 v16h;
typedef __attribute__((ext_vector_type(8)))  _Float16 v8h;
typedef __attribute__((ext_vector_type(8)))  float    v8f;

static constexpr int    Bsz   = 4;
static constexpr int    Cch   = 64;
static constexpr int    Hdim  = 128;
static constexpr int    Wdim  = 128;
static constexpr int    Lseq  = Hdim * Wdim;        // 16384
static constexpr size_t BL    = (size_t)Bsz * Lseq; // 65536
static constexpr int    DI    = 128;                // d_inner
static constexpr int    DS    = 16;                 // d_state
static constexpr int    DTR   = 4;                  // dt_rank
static constexpr int    XPN   = DTR + 2 * DS;       // 36
static constexpr int    CHUNK = 128;                // scan chunk length
static constexpr int    NCHNK = Lseq / CHUNK;       // 128 chunks per batch

// -------------------------- WMMA fragment helpers --------------------------

__device__ __forceinline__ v8f vzero8() {
    v8f a;
#pragma unroll
    for (int i = 0; i < 8; ++i) a[i] = 0.f;
    return a;
}

__device__ __forceinline__ v8f wmma32(v16h a, v16h b, v8f c) {
    // D = A(16x32 f16) * B(32x16 f16) + C(16x16 f32)
    return __builtin_amdgcn_wmma_f32_16x16x32_f16(
        /*neg_a=*/false, a, /*neg_b=*/false, b,
        /*c_mod=*/(short)0, c, /*reuse_a=*/false, /*reuse_b=*/false);
}

// A fragment, row-major A[M][K] (16-bit, 16x32 tile).
// ISA layout: lanes 0-15 hold M=lane, halves 0..7 = K k0+0..7, halves 8..15 =
// K k0+16..23; lanes 16-31 hold K k0+8..15 / k0+24..31.
__device__ __forceinline__ v16h load_a_frag(const h16_t* __restrict__ A,
                                            int lda, int m0, int k0, int lane) {
    int m   = m0 + (lane & 15);
    int off = (lane >> 4) << 3;
    const h16_t* p = A + (size_t)m * lda + (size_t)(k0 + off);
    v8h lo = *(const v8h*)p;
    v8h hi = *(const v8h*)(p + 16);
    v16h r;
#pragma unroll
    for (int j = 0; j < 8; ++j) { r[j] = lo[j]; r[j + 8] = hi[j]; }
    return r;
}

// B fragment from weight stored row-major W[N][K] (so B[k][n] = W[n][k]).
// ISA layout: lanes 0-15 hold N=lane, K k0+0..7 / k0+16..23; lanes 16-31 K k0+8..15 / k0+24..31.
// Two 16B loads per lane. Rows n >= N are zero (for ragged N).
__device__ __forceinline__ v16h load_b_frag(const h16_t* __restrict__ W,
                                            int ldb, int n0, int k0, int lane,
                                            int N) {
    int n  = n0 + (lane & 15);
    int kb = k0 + ((lane >> 4) << 3);
    if (n < N) {
        const h16_t* p = W + (size_t)n * ldb + (size_t)kb;
        v8h lo = *(const v8h*)p;
        v8h hi = *(const v8h*)(p + 16);
        v16h q;
#pragma unroll
        for (int j = 0; j < 8; ++j) { q[j] = lo[j]; q[j + 8] = hi[j]; }
        return q;
    }
    v16h r;
#pragma unroll
    for (int j = 0; j < 16; ++j) r[j] = (h16_t)0;
    return r;
}

// Implicit-GEMM patch chunk: 8 contiguous halves of the im2col row for pixel
// (b,y,x) from an NHWC f16 activation with (1<<CSH) channels; K ordering is
// k = (kh*3+kw)*(1<<CSH) + c. 8-aligned chunks never cross a tap boundary.
template <int CSH>
__device__ __forceinline__ void patch_chunk(v16h& r, int dst,
                                            const h16_t* __restrict__ src,
                                            int b, int y, int x, int k) {
    int tap = k >> CSH;
    int c   = k & ((1 << CSH) - 1);
    int yy  = y + (tap / 3) - 1;
    int xx  = x + (tap % 3) - 1;
    if ((unsigned)yy < 128u && (unsigned)xx < 128u) {
        v8h v = *(const v8h*)(src +
                  ((((size_t)b << 14) + ((size_t)yy << 7) + (size_t)xx) << CSH) + c);
#pragma unroll
        for (int j = 0; j < 8; ++j) r[dst + j] = v[j];
    } else {
#pragma unroll
        for (int j = 0; j < 8; ++j) r[dst + j] = (h16_t)0;
    }
}

// ------------------------------ weight prep --------------------------------

__global__ __launch_bounds__(256) void pack_f16(const float* __restrict__ s,
                                                h16_t* __restrict__ d, int n) {
    int g = blockIdx.x * 256 + threadIdx.x;
    if (g < n) d[g] = (h16_t)s[g];
}

// conv OIHW (O, I, 3, 3) -> [O][(kh*3+kw)*I + i] f16
__global__ __launch_bounds__(256) void pack_convw(const float* __restrict__ s,
                                                  h16_t* __restrict__ d,
                                                  int I, int total) {
    int g = blockIdx.x * 256 + threadIdx.x;
    if (g >= total) return;
    int o   = g / (I * 9);
    int rem = g - o * (I * 9);
    int tap = rem / I;
    int i   = rem - tap * I;
    d[g] = (h16_t)s[((size_t)(o * I + i)) * 9 + tap];
}

// ------------------------------ stage kernels ------------------------------

// LayerNorm over C=64 for each (b,l); also emit x in f16 into cat16 NHWC[c<64].
__global__ __launch_bounds__(256) void ln_kernel(const float* __restrict__ x,
                                                 const float* __restrict__ g,
                                                 const float* __restrict__ be,
                                                 h16_t* __restrict__ h16,
                                                 h16_t* __restrict__ cat16) {
    int bl = blockIdx.x * 256 + threadIdx.x;           // 0..65535
    int b  = bl >> 14;
    int l  = bl & (Lseq - 1);
    size_t base = ((size_t)b << 20) + (size_t)l;       // x[(b*64+c)*16384 + l]
    float s = 0.f, s2 = 0.f;
#pragma unroll 4
    for (int c = 0; c < 64; ++c) {
        float v = x[base + ((size_t)c << 14)];
        s += v; s2 += v * v;
    }
    float mu  = s * (1.f / 64.f);
    float var = s2 * (1.f / 64.f) - mu * mu;
    float inv = rsqrtf(var + 1e-5f);
#pragma unroll 4
    for (int c = 0; c < 64; ++c) {
        float v = x[base + ((size_t)c << 14)];
        h16[(size_t)bl * 64 + c]    = (h16_t)((v - mu) * inv * g[c] + be[c]);
        cat16[(size_t)bl * 128 + c] = (h16_t)v;
    }
}

// xz = h @ in_proj_w^T : M=65536, N=256, K=64.  cols<128 -> xm f32, else z16.
__global__ __launch_bounds__(256) void gemm_xz(const h16_t* __restrict__ A,
                                               const h16_t* __restrict__ W,
                                               float* __restrict__ xm,
                                               h16_t* __restrict__ z16) {
    int lane = threadIdx.x & 31;
    int wt   = blockIdx.x * 8 + (threadIdx.x >> 5);
    int m0   = (wt >> 2) << 4;
    int n0   = (wt & 3) << 6;
    v8f acc[4];
#pragma unroll
    for (int s = 0; s < 4; ++s) acc[s] = vzero8();
#pragma unroll
    for (int k0 = 0; k0 < 64; k0 += 32) {
        v16h a = load_a_frag(A, 64, m0, k0, lane);
#pragma unroll
        for (int s = 0; s < 4; ++s) {
            v16h bf = load_b_frag(W, 64, n0 + (s << 4), k0, lane, 256);
            acc[s]  = wmma32(a, bf, acc[s]);
        }
    }
    int cl = lane & 15, rb = (lane >> 4) << 3;
#pragma unroll
    for (int s = 0; s < 4; ++s) {
        int col = n0 + (s << 4) + cl;
#pragma unroll
        for (int i = 0; i < 8; ++i) {
            int   row = m0 + i + rb;
            float v   = acc[s][i];
            if (col < 128) xm[(size_t)row * 128 + col] = v;
            else           z16[(size_t)row * 128 + (col - 128)] = (h16_t)v;
        }
    }
}

// Depthwise causal conv1d (k=4) + SiLU -> xc f32 + xc16 f16.
__global__ __launch_bounds__(256) void dwconv_silu(const float* __restrict__ xm,
                                                   const float* __restrict__ w,
                                                   const float* __restrict__ bias,
                                                   float* __restrict__ xc,
                                                   h16_t* __restrict__ xc16) {
    size_t g  = (size_t)blockIdx.x * 256 + threadIdx.x; // BL*128
    int    d  = (int)(g & 127);
    size_t bl = g >> 7;
    int    l  = (int)(bl & (Lseq - 1));
    float acc = bias[d];
#pragma unroll
    for (int j = 0; j < 4; ++j) {
        int ll = l - 3 + j;
        if (ll >= 0) acc += xm[(bl - 3 + j) * 128 + d] * w[d * 4 + j];
    }
    float sv = acc * (1.f / (1.f + __expf(-acc)));
    xc[g]    = sv;
    xc16[g]  = (h16_t)sv;
}

// x_dbl = xc @ x_proj_w^T : M=65536, N=36, K=128 (wave covers 16x48, masked).
__global__ __launch_bounds__(256) void gemm_xdbl(const h16_t* __restrict__ A,
                                                 const h16_t* __restrict__ W,
                                                 float* __restrict__ xdbl) {
    int lane = threadIdx.x & 31;
    int wt   = blockIdx.x * 8 + (threadIdx.x >> 5);
    int m0   = wt << 4;
    v8f acc[3];
#pragma unroll
    for (int s = 0; s < 3; ++s) acc[s] = vzero8();
#pragma unroll
    for (int k0 = 0; k0 < 128; k0 += 32) {
        v16h a = load_a_frag(A, 128, m0, k0, lane);
#pragma unroll
        for (int s = 0; s < 3; ++s) {
            v16h bf = load_b_frag(W, 128, s << 4, k0, lane, XPN);
            acc[s]  = wmma32(a, bf, acc[s]);
        }
    }
    int cl = lane & 15, rb = (lane >> 4) << 3;
#pragma unroll
    for (int s = 0; s < 3; ++s) {
        int col = (s << 4) + cl;
        if (col < XPN) {
#pragma unroll
            for (int i = 0; i < 8; ++i)
                xdbl[(size_t)(m0 + i + rb) * XPN + col] = acc[s][i];
        }
    }
}

// dt = softplus(dtr @ dt_proj_w^T + dt_proj_b)
__global__ __launch_bounds__(256) void dtproj(const float* __restrict__ xdbl,
                                              const float* __restrict__ dtw,
                                              const float* __restrict__ dtb,
                                              float* __restrict__ dt) {
    size_t g  = (size_t)blockIdx.x * 256 + threadIdx.x; // BL*128
    int    d  = (int)(g & 127);
    size_t bl = g >> 7;
    float v = dtb[d];
#pragma unroll
    for (int r = 0; r < DTR; ++r) v += xdbl[bl * XPN + r] * dtw[d * DTR + r];
    dt[g] = (v > 20.f) ? v : log1pf(__expf(v));
}

// Scan pass 1: per (b,chunk,d) compute chunk-final affine (P = prod dA, q).
__global__ __launch_bounds__(128) void scan_chunk(const float* __restrict__ dt,
                                                  const float* __restrict__ xc,
                                                  const float* __restrict__ xdbl,
                                                  const float* __restrict__ A_log,
                                                  float* __restrict__ Pc,
                                                  float* __restrict__ qc) {
    __shared__ float Bsh[CHUNK * DS];
    int d  = threadIdx.x;                  // 0..127
    int bc = blockIdx.x;                   // b*128 + chunk
    size_t bl0 = ((size_t)(bc >> 7) << 14) + (size_t)(bc & 127) * CHUNK;

    float Ar[DS];
#pragma unroll
    for (int s = 0; s < DS; ++s) Ar[s] = -__expf(A_log[d * DS + s]);

#pragma unroll
    for (int s = 0; s < DS; ++s)
        Bsh[d * DS + s] = xdbl[(bl0 + d) * XPN + DTR + s];
    __syncthreads();

    float P[DS], q[DS];
#pragma unroll
    for (int s = 0; s < DS; ++s) { P[s] = 1.f; q[s] = 0.f; }

    for (int t = 0; t < CHUNK; ++t) {
        size_t bl  = bl0 + t;
        float  dtv = dt[bl * 128 + d];
        float  xv  = xc[bl * 128 + d];
        if (t + 16 < CHUNK) {
            __builtin_prefetch(&dt[(bl + 16) * 128 + d], 0, 0);
            __builtin_prefetch(&xc[(bl + 16) * 128 + d], 0, 0);
        }
        float u = dtv * xv;
#pragma unroll
        for (int s = 0; s < DS; ++s) {
            float a = __expf(dtv * Ar[s]);
            q[s] = q[s] * a + u * Bsh[t * DS + s];
            P[s] *= a;
        }
    }
    size_t base = (((size_t)bc) * 128 + d) * DS;
#pragma unroll
    for (int s = 0; s < DS; ++s) { Pc[base + s] = P[s]; qc[base + s] = q[s]; }
}

// Scan pass 2: tiny sequential combine over chunks -> chunk-entry states h0.
__global__ __launch_bounds__(256) void scan_combine(const float* __restrict__ Pc,
                                                    const float* __restrict__ qc,
                                                    float* __restrict__ h0) {
    int g = blockIdx.x * 256 + threadIdx.x;   // 4*128*16 = 8192
    int s = g & 15, d = (g >> 4) & 127, b = g >> 11;
    float h = 0.f;
    for (int c = 0; c < NCHNK; ++c) {
        size_t idx = ((((size_t)b * NCHNK + c) * 128 + d) * DS) + s;
        h0[idx] = h;
        h = Pc[idx] * h + qc[idx];
    }
}

// Scan pass 3: re-apply with true entry states; fuse +xc*Dp and *SiLU(z) -> y16.
__global__ __launch_bounds__(128) void scan_apply(const float* __restrict__ dt,
                                                  const float* __restrict__ xc,
                                                  const float* __restrict__ xdbl,
                                                  const float* __restrict__ A_log,
                                                  const float* __restrict__ h0b,
                                                  const float* __restrict__ Dp,
                                                  const h16_t* __restrict__ z16,
                                                  h16_t* __restrict__ y16) {
    __shared__ float Bsh[CHUNK * DS];
    __shared__ float Csh[CHUNK * DS];
    int d  = threadIdx.x;
    int bc = blockIdx.x;
    size_t bl0 = ((size_t)(bc >> 7) << 14) + (size_t)(bc & 127) * CHUNK;

    float Ar[DS];
#pragma unroll
    for (int s = 0; s < DS; ++s) Ar[s] = -__expf(A_log[d * DS + s]);

#pragma unroll
    for (int s = 0; s < DS; ++s) {
        Bsh[d * DS + s] = xdbl[(bl0 + d) * XPN + DTR + s];
        Csh[d * DS + s] = xdbl[(bl0 + d) * XPN + DTR + DS + s];
    }
    __syncthreads();

    float h[DS];
    size_t base = (((size_t)bc) * 128 + d) * DS;
#pragma unroll
    for (int s = 0; s < DS; ++s) h[s] = h0b[base + s];
    float Dv = Dp[d];

    for (int t = 0; t < CHUNK; ++t) {
        size_t bl  = bl0 + t;
        float  dtv = dt[bl * 128 + d];
        float  xv  = xc[bl * 128 + d];
        if (t + 16 < CHUNK) {
            __builtin_prefetch(&dt[(bl + 16) * 128 + d], 0, 0);
            __builtin_prefetch(&xc[(bl + 16) * 128 + d], 0, 0);
        }
        float u = dtv * xv;
        float y = 0.f;
#pragma unroll
        for (int s = 0; s < DS; ++s) {
            float a = __expf(dtv * Ar[s]);
            h[s] = h[s] * a + u * Bsh[t * DS + s];
            y += h[s] * Csh[t * DS + s];
        }
        float yy   = y + xv * Dv;
        float zv   = (float)z16[bl * 128 + d];
        float gate = zv * (1.f / (1.f + __expf(-zv)));
        y16[bl * 128 + d] = (h16_t)(yy * gate);
    }
}

// m_out = y @ out_proj_w^T : M=65536, N=64, K=128 -> cat16 NHWC channels 64..127.
__global__ __launch_bounds__(256) void gemm_mout(const h16_t* __restrict__ A,
                                                 const h16_t* __restrict__ W,
                                                 h16_t* __restrict__ cat16) {
    int lane = threadIdx.x & 31;
    int wt   = blockIdx.x * 8 + (threadIdx.x >> 5);
    int m0   = wt << 4;
    v8f acc[4];
#pragma unroll
    for (int s = 0; s < 4; ++s) acc[s] = vzero8();
#pragma unroll
    for (int k0 = 0; k0 < 128; k0 += 32) {
        v16h a = load_a_frag(A, 128, m0, k0, lane);
#pragma unroll
        for (int s = 0; s < 4; ++s) {
            v16h bf = load_b_frag(W, 128, s << 4, k0, lane, 64);
            acc[s]  = wmma32(a, bf, acc[s]);
        }
    }
    int cl = lane & 15, rb = (lane >> 4) << 3;
#pragma unroll
    for (int s = 0; s < 4; ++s) {
        int col = (s << 4) + cl;
#pragma unroll
        for (int i = 0; i < 8; ++i)
            cat16[(size_t)(m0 + i + rb) * 128 + 64 + col] = (h16_t)acc[s][i];
    }
}

// conv1 3x3 (128->64) + relu as implicit GEMM: M=65536, N=64, K=1152 -> t1 f16.
__global__ __launch_bounds__(256) void conv1_wmma(const h16_t* __restrict__ cat16,
                                                  const h16_t* __restrict__ w1r,
                                                  h16_t* __restrict__ t1) {
    int lane = threadIdx.x & 31;
    int wt   = blockIdx.x * 8 + (threadIdx.x >> 5);
    int m0   = wt << 4;
    int m    = m0 + (lane & 15);
    int b    = m >> 14, y = (m >> 7) & 127, x = m & 127;
    int off  = (lane >> 4) << 3;
    v8f acc[4];
#pragma unroll
    for (int s = 0; s < 4; ++s) acc[s] = vzero8();
    for (int k0 = 0; k0 < 1152; k0 += 32) {
        v16h a;
        patch_chunk<7>(a, 0, cat16, b, y, x, k0 + off);
        patch_chunk<7>(a, 8, cat16, b, y, x, k0 + off + 16);
#pragma unroll
        for (int s = 0; s < 4; ++s) {
            v16h bf = load_b_frag(w1r, 1152, s << 4, k0, lane, 64);
            acc[s]  = wmma32(a, bf, acc[s]);
        }
    }
    int cl = lane & 15, rb = (lane >> 4) << 3;
#pragma unroll
    for (int s = 0; s < 4; ++s) {
        int col = (s << 4) + cl;
#pragma unroll
        for (int i = 0; i < 8; ++i)
            t1[(size_t)(m0 + i + rb) * 64 + col] = (h16_t)fmaxf(acc[s][i], 0.f);
    }
}

// conv2 3x3 (64->64) + skip 1x1 (128->64) + relu, fused -> d_out NCHW f32.
__global__ __launch_bounds__(256) void conv2skip_wmma(const h16_t* __restrict__ t1,
                                                      const h16_t* __restrict__ cat16,
                                                      const h16_t* __restrict__ w2r,
                                                      const h16_t* __restrict__ wsk,
                                                      float* __restrict__ out) {
    int lane = threadIdx.x & 31;
    int wt   = blockIdx.x * 8 + (threadIdx.x >> 5);
    int m0   = wt << 4;
    int m    = m0 + (lane & 15);
    int b    = m >> 14, y = (m >> 7) & 127, x = m & 127;
    int off  = (lane >> 4) << 3;
    v8f acc[4];
#pragma unroll
    for (int s = 0; s < 4; ++s) acc[s] = vzero8();
    // conv2 over t1 (NHWC, 64 ch): K = 576
    for (int k0 = 0; k0 < 576; k0 += 32) {
        v16h a;
        patch_chunk<6>(a, 0, t1, b, y, x, k0 + off);
        patch_chunk<6>(a, 8, t1, b, y, x, k0 + off + 16);
#pragma unroll
        for (int s = 0; s < 4; ++s) {
            v16h bf = load_b_frag(w2r, 576, s << 4, k0, lane, 64);
            acc[s]  = wmma32(a, bf, acc[s]);
        }
    }
    // skip 1x1 over cat16: K = 128 (plain row-major GEMM accumulate)
#pragma unroll
    for (int k0 = 0; k0 < 128; k0 += 32) {
        v16h a = load_a_frag(cat16, 128, m0, k0, lane);
#pragma unroll
        for (int s = 0; s < 4; ++s) {
            v16h bf = load_b_frag(wsk, 128, s << 4, k0, lane, 64);
            acc[s]  = wmma32(a, bf, acc[s]);
        }
    }
    int cl = lane & 15, rb = (lane >> 4) << 3;
#pragma unroll
    for (int s = 0; s < 4; ++s) {
        int col = (s << 4) + cl;  // output channel o
#pragma unroll
        for (int i = 0; i < 8; ++i) {
            int row = m0 + i + rb;                  // pixel index b*L + y*W + x
            int bb  = row >> 14;
            int pix = row & (Lseq - 1);
            out[(((size_t)bb * 64 + col) << 14) + pix] = fmaxf(acc[s][i], 0.f);
        }
    }
}

// ------------------------------ host launcher ------------------------------

extern "C" void kernel_launch(void* const* d_in, const int* in_sizes, int n_in,
                              void* d_out, int out_size, void* d_ws, size_t ws_size,
                              hipStream_t stream) {
    (void)in_sizes; (void)n_in; (void)out_size; (void)ws_size;

    const float* x        = (const float*)d_in[0];
    const float* ln_g     = (const float*)d_in[1];
    const float* ln_b     = (const float*)d_in[2];
    const float* in_projw = (const float*)d_in[3];
    const float* conv1d_w = (const float*)d_in[4];
    const float* conv1d_b = (const float*)d_in[5];
    const float* x_projw  = (const float*)d_in[6];
    const float* dt_projw = (const float*)d_in[7];
    const float* dt_projb = (const float*)d_in[8];
    const float* A_log    = (const float*)d_in[9];
    const float* Dp       = (const float*)d_in[10];
    const float* out_projw= (const float*)d_in[11];
    const float* skip_w   = (const float*)d_in[12];
    const float* conv1_w  = (const float*)d_in[13];
    const float* conv2_w  = (const float*)d_in[14];

    // workspace arena (256B-aligned slices); total ~157 MB with aliasing
    uint8_t* wsb = (uint8_t*)d_ws;
    size_t   off = 0;
    auto alloc = [&](size_t bytes) -> void* {
        void* p = wsb + off;
        off = (off + bytes + 255) & ~(size_t)255;
        return p;
    };
    constexpr int SZ_WIP = 256 * 64;
    constexpr int SZ_WXP = XPN * 128;
    constexpr int SZ_WOP = 64 * 128;
    constexpr int SZ_WSK = 64 * 128;
    constexpr int SZ_W1R = 64 * 1152;
    constexpr int SZ_W2R = 64 * 576;

    h16_t* w_ip  = (h16_t*)alloc((size_t)SZ_WIP * 2);
    h16_t* w_xp  = (h16_t*)alloc((size_t)SZ_WXP * 2);
    h16_t* w_op  = (h16_t*)alloc((size_t)SZ_WOP * 2);
    h16_t* w_sk  = (h16_t*)alloc((size_t)SZ_WSK * 2);
    h16_t* w_1r  = (h16_t*)alloc((size_t)SZ_W1R * 2);
    h16_t* w_2r  = (h16_t*)alloc((size_t)SZ_W2R * 2);
    h16_t* h16b  = (h16_t*)alloc(BL * 64 * 2);
    float* xm    = (float*)alloc(BL * 128 * 4);
    h16_t* z16   = (h16_t*)alloc(BL * 128 * 2);
    float* xc    = (float*)alloc(BL * 128 * 4);
    h16_t* xc16  = (h16_t*)alloc(BL * 128 * 2);
    float* xdbl  = (float*)alloc(BL * XPN * 4);
    float* Pc    = (float*)alloc((size_t)Bsz * NCHNK * 128 * DS * 4);
    float* qc    = (float*)alloc((size_t)Bsz * NCHNK * 128 * DS * 4);
    float* h0b   = (float*)alloc((size_t)Bsz * NCHNK * 128 * DS * 4);
    h16_t* cat16 = (h16_t*)alloc(BL * 128 * 2);
    h16_t* t1    = (h16_t*)alloc(BL * 64 * 2);
    float* dtbuf = xm;    // xm dead after dwconv_silu
    h16_t* y16   = xc16;  // xc16 dead after gemm_xdbl

    // weight prep
    pack_f16<<<(SZ_WIP + 255) / 256, 256, 0, stream>>>(in_projw, w_ip, SZ_WIP);
    pack_f16<<<(SZ_WXP + 255) / 256, 256, 0, stream>>>(x_projw, w_xp, SZ_WXP);
    pack_f16<<<(SZ_WOP + 255) / 256, 256, 0, stream>>>(out_projw, w_op, SZ_WOP);
    pack_f16<<<(SZ_WSK + 255) / 256, 256, 0, stream>>>(skip_w, w_sk, SZ_WSK);
    pack_convw<<<(SZ_W1R + 255) / 256, 256, 0, stream>>>(conv1_w, w_1r, 128, SZ_W1R);
    pack_convw<<<(SZ_W2R + 255) / 256, 256, 0, stream>>>(conv2_w, w_2r, 64, SZ_W2R);

    // mamba path
    ln_kernel<<<(int)(BL / 256), 256, 0, stream>>>(x, ln_g, ln_b, h16b, cat16);
    gemm_xz<<<(int)(BL / 16 * 4 / 8), 256, 0, stream>>>(h16b, w_ip, xm, z16);
    dwconv_silu<<<(int)(BL * 128 / 256), 256, 0, stream>>>(xm, conv1d_w, conv1d_b, xc, xc16);
    gemm_xdbl<<<(int)(BL / 16 / 8), 256, 0, stream>>>(xc16, w_xp, xdbl);
    dtproj<<<(int)(BL * 128 / 256), 256, 0, stream>>>(xdbl, dt_projw, dt_projb, dtbuf);

    // chunked parallel scan (3 passes)
    scan_chunk<<<Bsz * NCHNK, 128, 0, stream>>>(dtbuf, xc, xdbl, A_log, Pc, qc);
    scan_combine<<<(Bsz * 128 * DS) / 256, 256, 0, stream>>>(Pc, qc, h0b);
    scan_apply<<<Bsz * NCHNK, 128, 0, stream>>>(dtbuf, xc, xdbl, A_log, h0b, Dp, z16, y16);

    // output projection + CNN tail
    gemm_mout<<<(int)(BL / 16 / 8), 256, 0, stream>>>(y16, w_op, cat16);
    conv1_wmma<<<(int)(BL / 16 / 8), 256, 0, stream>>>(cat16, w_1r, t1);
    conv2skip_wmma<<<(int)(BL / 16 / 8), 256, 0, stream>>>(t1, cat16, w_2r, w_sk,
                                                           (float*)d_out);
}
